// E_MHSA_4518305595980
// MI455X (gfx1250) — compile-verified, matrix operand order checked by reading, not run on hardware
//
#include <hip/hip_runtime.h>
#include <hip/hip_bf16.h>

// ---------------------------------------------------------------------------
// E-MHSA (PVT SRA attention) for gfx1250 / MI455X.
// B=4, N=3136, C=384, H=12, d=32, SR pool=4 -> Nk=784 (padded to 800).
// All GEMMs use v_wmma_f32_16x16x32_f16 (wave32). Attention is fused per
// (b,h,qblock): K and V^T heads live entirely in LDS (~106KB of the 320KB WGP
// LDS), staged with global_load_async_to_lds_b128 (ASYNCcnt); softmax is an
// exact two-pass with scale*log2e folded into Q so the inner loops are pure
// wmma + max / exp2; the attention matrix never reaches HBM.
// ---------------------------------------------------------------------------

typedef __attribute__((ext_vector_type(16))) _Float16 v16h;
typedef __attribute__((ext_vector_type(8)))  float    v8f;

#define CC   384
#define HH   12
#define DD   32
#define NN   3136
#define NKK  784
#define NKP  800
#define BB   4

// scale(1/sqrt(32)) * log2(e): folded into Q so softmax uses exp2 directly.
#define QSCALE 0.25508680714244825f

union FragU { uint4 u[2]; v16h h; };

// Load a 16x32 f16 WMMA A/B fragment slice for this lane from a row-major row.
// Lane group g (lane>>4): halves at K = [8g .. 8g+7] and [16+8g .. 23+8g].
__device__ inline v16h load_frag(const _Float16* row, int g) {
    FragU f;
    f.u[0] = *reinterpret_cast<const uint4*>(row + 8 * g);
    f.u[1] = *reinterpret_cast<const uint4*>(row + 16 + 8 * g);
    return f.h;
}

__device__ inline v8f vzero8() {
    v8f c;
#pragma unroll
    for (int i = 0; i < 8; ++i) c[i] = 0.0f;
    return c;
}

__device__ inline v8f wmma_f16(v16h a, v16h b, v8f c) {
    return __builtin_amdgcn_wmma_f32_16x16x32_f16(
        false, a, false, b, (short)0, c, false, false);
}

// CDNA5 async memory->LDS copy (16B per active lane), tracked by ASYNCcnt.
__device__ inline void async_copy_b128(unsigned ldsAddr, unsigned gOff,
                                       unsigned long long gBase) {
    asm volatile("global_load_async_to_lds_b128 %0, %1, %2"
                 :: "v"(ldsAddr), "v"(gOff), "s"(gBase) : "memory");
}
__device__ inline void wait_asynccnt0() {
    asm volatile("s_wait_asynccnt 0x0" ::: "memory");
}

// ---------------------------------------------------------------------------
// Prep kernels
// ---------------------------------------------------------------------------
__global__ void cvt_weights(const float* __restrict__ wq, const float* __restrict__ wk,
                            const float* __restrict__ wv, const float* __restrict__ wp,
                            _Float16* __restrict__ oq, _Float16* __restrict__ ok,
                            _Float16* __restrict__ ov, _Float16* __restrict__ op, int n) {
    int i = blockIdx.x * blockDim.x + threadIdx.x;
    if (i < n) {
        oq[i] = (_Float16)wq[i];
        ok[i] = (_Float16)wk[i];
        ov[i] = (_Float16)wv[i];
        op[i] = (_Float16)wp[i];
    }
}

__global__ void cvt_x(const float* __restrict__ x, _Float16* __restrict__ xh, int n) {
    int i = blockIdx.x * blockDim.x + threadIdx.x;
    if (i < n) xh[i] = (_Float16)x[i];
}

// avg-pool over 4 tokens + inference BatchNorm -> f16 [B*Nk, C]
__global__ void pool_bn(const float* __restrict__ x,
                        const float* __restrict__ mean, const float* __restrict__ var,
                        const float* __restrict__ gamma, const float* __restrict__ beta,
                        _Float16* __restrict__ out, int total) {
    int i = blockIdx.x * blockDim.x + threadIdx.x;
    if (i >= total) return;
    int c  = i % CC;
    int r  = i / CC;
    int nk = r % NKK;
    int b  = r / NKK;
    const float* p = x + ((size_t)b * NN + (size_t)nk * 4) * CC + c;
    float m = 0.25f * (p[0] + p[CC] + p[2 * CC] + p[3 * CC]);
    float y = (m - mean[c]) * rsqrtf(var[c] + 1e-5f) * gamma[c] + beta[c];
    out[i] = (_Float16)y;
}

// ---------------------------------------------------------------------------
// WMMA GEMM: out[m][o] = (sum_i A[m][i] * W[o][i] + bias[o]) * outScale
//   A: f16 row-major [M, 384]; W: f16 row-major [384, 384] (rows are the WMMA
//   B fragments since the GEMM is x @ W^T).
// One wave -> 16(M) x 64(N) tile; 4 waves/block; grid = (Mtiles/4, 384/64).
// MODE 0: f16 out, per-head layout  [B, H, rowsPerB, 32]   (q, k)
// MODE 1: f16 out, transposed head  [B, H, 32, NKP]        (v^T)
// MODE 2: f32 out, plain row-major  [M, 384]               (final projection)
// ---------------------------------------------------------------------------
template <int MODE>
__global__ void gemm_wmma(const _Float16* __restrict__ A, const _Float16* __restrict__ W,
                          const float* __restrict__ bias, void* __restrict__ Out,
                          int M, int rowsPerB, float outScale) {
    int lane = threadIdx.x & 31;
    int wave = threadIdx.x >> 5;
    int mt = blockIdx.x * 4 + wave;
    if (mt * 16 >= M) return;               // wave-uniform: EXEC stays all-ones
    int n0 = blockIdx.y * 64;
    int g  = lane >> 4;
    int lm = lane & 15;

    const _Float16* arow = A + (size_t)(mt * 16 + lm) * CC;
    __builtin_prefetch(arow, 0, 1);         // global_prefetch_b8

    v8f c0 = vzero8(), c1 = vzero8(), c2 = vzero8(), c3 = vzero8();
#pragma unroll 4
    for (int kt = 0; kt < CC / 32; ++kt) {
        v16h a  = load_frag(arow + kt * 32, g);
        v16h b0 = load_frag(W + (size_t)(n0 +  0 + lm) * CC + kt * 32, g);
        v16h b1 = load_frag(W + (size_t)(n0 + 16 + lm) * CC + kt * 32, g);
        v16h b2 = load_frag(W + (size_t)(n0 + 32 + lm) * CC + kt * 32, g);
        v16h b3 = load_frag(W + (size_t)(n0 + 48 + lm) * CC + kt * 32, g);
        c0 = wmma_f16(a, b0, c0);
        c1 = wmma_f16(a, b1, c1);
        c2 = wmma_f16(a, b2, c2);
        c3 = wmma_f16(a, b3, c3);
    }

    v8f acc[4] = {c0, c1, c2, c3};
#pragma unroll
    for (int t = 0; t < 4; ++t) {
        int o = n0 + t * 16 + lm;
        float bi = bias[o];
#pragma unroll
        for (int j = 0; j < 8; ++j) {
            float val = (acc[t][j] + bi) * outScale;
            int m = mt * 16 + j + 8 * g;
            if (MODE == 2) {
                ((float*)Out)[(size_t)m * CC + o] = val;
            } else {
                int b = m / rowsPerB, n = m % rowsPerB;
                int h = o >> 5, dc = o & 31;
                size_t idx;
                if (MODE == 0) idx = ((size_t)(b * HH + h) * rowsPerB + n) * DD + dc;
                else           idx = ((size_t)(b * HH + h) * DD + dc) * NKP + n;
                ((_Float16*)Out)[idx] = (_Float16)val;
            }
        }
    }
}

// ---------------------------------------------------------------------------
// Fused attention: per (qblock64, h, b) workgroup of 4 waves; each wave owns
// 16 query rows. K head [Nk,32] and V^T head [32,NKP] staged in LDS via
// async-to-LDS DMA; exact softmax over Nk in two passes (max, then exp2+PV).
// Q already carries scale*log2e. P goes through a per-wave LDS staging tile
// to convert WMMA C layout -> A layout.
// Output written f16 row-major [B*N, C] (channel = h*32 + d) for final GEMM.
// ---------------------------------------------------------------------------
__global__ void attn_kernel(const _Float16* __restrict__ qh,
                            const _Float16* __restrict__ kh,
                            const _Float16* __restrict__ vth,
                            _Float16* __restrict__ attno) {
    __shared__ __align__(16) _Float16 Klds[NKP * DD];      // 51200 B
    __shared__ __align__(16) _Float16 Vlds[DD * NKP];      // 51200 B
    __shared__ __align__(16) _Float16 stage[4][16 * 32];   //  4096 B

    int b = blockIdx.z, h = blockIdx.y;
    int tid  = threadIdx.x;
    int lane = tid & 31;
    int wave = tid >> 5;
    int g  = lane >> 4;
    int lm = lane & 15;

    // --- stage K and V^T heads into LDS via async DMA (ASYNCcnt) ---
    unsigned long long ksrc =
        (unsigned long long)(uintptr_t)(kh  + (size_t)(b * HH + h) * NKK * DD);
    unsigned long long vsrc =
        (unsigned long long)(uintptr_t)(vth + (size_t)(b * HH + h) * DD * NKP);
    unsigned kbase = (unsigned)(uintptr_t)&Klds[0];
    unsigned vbase = (unsigned)(uintptr_t)&Vlds[0];
    for (int i = tid; i < NKK * DD / 8; i += 128)          // 3136 x 16B
        async_copy_b128(kbase + i * 16, i * 16, ksrc);
    for (int i = tid; i < DD * NKP / 8; i += 128)          // 3200 x 16B
        async_copy_b128(vbase + i * 16, i * 16, vsrc);
    wait_asynccnt0();
    // zero the K padding rows and the V^T key-padding columns (784..799)
    uint4 z = make_uint4(0, 0, 0, 0);
    for (int i = tid; i < (NKP - NKK) * DD / 8; i += 128)
        ((uint4*)Klds)[NKK * DD / 8 + i] = z;
    for (int i = tid; i < DD * (NKP - NKK); i += 128)
        Vlds[(i / 16) * NKP + NKK + (i % 16)] = (_Float16)0.0f;
    __syncthreads();

    int q0 = blockIdx.x * 64 + wave * 16;
    const _Float16* qrow = qh + ((size_t)(b * HH + h) * NN + q0 + lm) * DD;
    v16h qa = load_frag(qrow, g);              // A fragment: rows q, K = d

    // --- pass 1: row max of S' = (scale*log2e) * Q K^T (scale is inside Q) ---
    float vm[8];
#pragma unroll
    for (int j = 0; j < 8; ++j) vm[j] = -1e30f;
    for (int t = 0; t < NKP / 16; ++t) {
        v16h bk = load_frag(Klds + (size_t)(t * 16 + lm) * DD, g);
        v8f s = wmma_f16(qa, bk, vzero8());
#pragma unroll
        for (int j = 0; j < 8; ++j) vm[j] = fmaxf(vm[j], s[j]);
    }
#pragma unroll
    for (int j = 0; j < 8; ++j)
        for (int off = 1; off < 16; off <<= 1)
            vm[j] = fmaxf(vm[j], __shfl_xor(vm[j], off, 32));

    // --- pass 2: P = exp2(S' - m); l += rowsum(P); acc += P @ V ---
    float l[8];
#pragma unroll
    for (int j = 0; j < 8; ++j) l[j] = 0.0f;
    v8f o0 = vzero8(), o1 = vzero8();
    _Float16* st = stage[wave];

    for (int kb = 0; kb < NKP / 32; ++kb) {
#pragma unroll
        for (int t = 0; t < 2; ++t) {
            int key = kb * 32 + t * 16 + lm;
            v16h bk = load_frag(Klds + (size_t)key * DD, g);
            v8f s = wmma_f16(qa, bk, vzero8());
            float msk = (key < NKK) ? 1.0f : 0.0f;
#pragma unroll
            for (int j = 0; j < 8; ++j) {
                float p = msk * exp2f(s[j] - vm[j]);
                l[j] += p;
                st[(j + 8 * g) * 32 + t * 16 + lm] = (_Float16)p;
            }
        }
        __builtin_amdgcn_wave_barrier();
        v16h pa  = load_frag(st + lm * 32, g);                          // P as A frag
        v16h bv0 = load_frag(Vlds + (size_t)(lm)      * NKP + kb * 32, g);
        v16h bv1 = load_frag(Vlds + (size_t)(16 + lm) * NKP + kb * 32, g);
        o0 = wmma_f16(pa, bv0, o0);
        o1 = wmma_f16(pa, bv1, o1);
        __builtin_amdgcn_wave_barrier();
    }

#pragma unroll
    for (int j = 0; j < 8; ++j)
        for (int off = 1; off < 16; off <<= 1)
            l[j] += __shfl_xor(l[j], off, 32);

#pragma unroll
    for (int j = 0; j < 8; ++j) {
        float inv = 1.0f / l[j];
        size_t base = ((size_t)b * NN + q0 + j + 8 * g) * CC + h * DD;
        attno[base + lm]      = (_Float16)(o0[j] * inv);
        attno[base + 16 + lm] = (_Float16)(o1[j] * inv);
    }
}

// ---------------------------------------------------------------------------
// Host launch
// ---------------------------------------------------------------------------
extern "C" void kernel_launch(void* const* d_in, const int* in_sizes, int n_in,
                              void* d_out, int out_size, void* d_ws, size_t ws_size,
                              hipStream_t stream) {
    const float* x     = (const float*)d_in[0];
    const float* Wq    = (const float*)d_in[1];
    const float* bq    = (const float*)d_in[2];
    const float* Wk    = (const float*)d_in[3];
    const float* bk    = (const float*)d_in[4];
    const float* Wv    = (const float*)d_in[5];
    const float* bv    = (const float*)d_in[6];
    const float* Wp    = (const float*)d_in[7];
    const float* bp    = (const float*)d_in[8];
    const float* gamma = (const float*)d_in[9];
    const float* beta  = (const float*)d_in[10];
    const float* mean  = (const float*)d_in[11];
    const float* var   = (const float*)d_in[12];
    float* out = (float*)d_out;

    // workspace layout (all 256B-aligned)
    char* ws = (char*)d_ws;
    size_t off = 0;
    _Float16* xh    = (_Float16*)(ws + off); off += (size_t)BB * NN * CC * 2;
    _Float16* xph   = (_Float16*)(ws + off); off += (size_t)BB * NKK * CC * 2;
    _Float16* Wqh   = (_Float16*)(ws + off); off += (size_t)CC * CC * 2;
    _Float16* Wkh   = (_Float16*)(ws + off); off += (size_t)CC * CC * 2;
    _Float16* Wvh   = (_Float16*)(ws + off); off += (size_t)CC * CC * 2;
    _Float16* Wph   = (_Float16*)(ws + off); off += (size_t)CC * CC * 2;
    _Float16* qh    = (_Float16*)(ws + off); off += (size_t)BB * HH * NN * DD * 2;
    _Float16* kh    = (_Float16*)(ws + off); off += (size_t)BB * HH * NKK * DD * 2;
    _Float16* vth   = (_Float16*)(ws + off); off += (size_t)BB * HH * DD * NKP * 2;
    _Float16* attno = (_Float16*)(ws + off); off += (size_t)BB * NN * CC * 2;

    const int nW = CC * CC;                       // 147456
    const int nX = BB * NN * CC;                  // 4,816,896
    const int nP = BB * NKK * CC;                 // 1,204,224

    cvt_weights<<<(nW + 255) / 256, 256, 0, stream>>>(Wq, Wk, Wv, Wp, Wqh, Wkh, Wvh, Wph, nW);
    cvt_x<<<(nX + 255) / 256, 256, 0, stream>>>(x, xh, nX);
    pool_bn<<<(nP + 255) / 256, 256, 0, stream>>>(x, mean, var, gamma, beta, xph, nP);

    // q = (x @ Wq^T + bq) * scale*log2e   -> [B,H,N,32] f16
    gemm_wmma<0><<<dim3((BB * NN / 16) / 4, CC / 64), 128, 0, stream>>>(
        xh, Wqh, bq, (void*)qh, BB * NN, NN, QSCALE);
    // k = x_ @ Wk^T + bk  -> [B,H,Nk,32] f16
    gemm_wmma<0><<<dim3((BB * NKK / 16) / 4, CC / 64), 128, 0, stream>>>(
        xph, Wkh, bk, (void*)kh, BB * NKK, NKK, 1.0f);
    // v = x_ @ Wv^T + bv  -> [B,H,32,NKP] f16 (transposed per head)
    gemm_wmma<1><<<dim3((BB * NKK / 16) / 4, CC / 64), 128, 0, stream>>>(
        xph, Wvh, bv, (void*)vth, BB * NKK, NKK, 1.0f);

    // fused softmax2(Q'K^T) @ V  -> attno f16 [B*N, C]
    attn_kernel<<<dim3(NN / 64, HH, BB), 128, 0, stream>>>(qh, kh, vth, attno);

    // final projection: out = attno @ Wp^T + bp  -> f32 [B*N, C]
    gemm_wmma<2><<<dim3((BB * NN / 16) / 4, CC / 64), 128, 0, stream>>>(
        attno, Wph, bp, (void*)out, BB * NN, 0, 1.0f);
}
